// MambaTTSDecoderLayer_65017214927439
// MI455X (gfx1250) — matrix operand order repaired and checked by reading, not compiled, not run on hardware
//
#include <hip/hip_runtime.h>
#include <math.h>

// ---------------- problem constants ----------------
#define D_MODEL 1024
#define N_HEADS 16
#define HEAD_D  64
#define D_FF    4096
#define D_STYLE 256
#define D_INNER 2048
#define D_STATE 16
#define D_CONV  4
#define DT_RANK 64
#define BATCH   2
#define SEQ_L   512
#define SEQ_T   256
#define ROWS_X  (BATCH*SEQ_L)   // 1024
#define ROWS_T  (BATCH*SEQ_T)   // 512

typedef unsigned short u16;
typedef __attribute__((ext_vector_type(16))) __bf16 bf16x16;
typedef __attribute__((ext_vector_type(8)))  float  f32x8;

union FragBF {      // 16 bf16 = 32 bytes = two int4 LDS/global loads
    bf16x16 v;
    int4    q[2];
};

static __device__ __forceinline__ u16 f2bf(float f) {
    unsigned u = __builtin_bit_cast(unsigned, f);
    unsigned r = u + 0x7FFFu + ((u >> 16) & 1u);   // round-to-nearest-even
    return (u16)(r >> 16);
}

// CDNA5 async global->LDS copy (no VGPR round trip, tracked by ASYNCcnt).
// VDST vgpr = wave-relative LDS byte offset (low 32 bits of generic pointer),
// VADDR = 64-bit global address pair, no saddr.
#define ASYNC_B128(ldsp, gp)                                                   \
    asm volatile("global_load_async_to_lds_b128 %0, %1, off"                   \
                 :: "v"((unsigned)(unsigned long long)(ldsp)),                 \
                    "v"((unsigned long long)(gp)) : "memory")
#define WAIT_ASYNCCNT(n)                                                       \
    asm volatile("s_wait_asynccnt " #n ::: "memory")

// ---------------- fp32 -> bf16 convert ----------------
__global__ __launch_bounds__(256) void cvt_kernel(const float* __restrict__ src,
                                                  u16* __restrict__ dst, int n) {
    int i = blockIdx.x * blockDim.x + threadIdx.x;
    int stride = gridDim.x * blockDim.x;
    for (; i < n; i += stride) dst[i] = f2bf(src[i]);
}

// extract dt (first 64 of 96 cols of xdbl) as bf16
__global__ __launch_bounds__(256) void dtcvt_kernel(const float* __restrict__ xdbl,
                                                    u16* __restrict__ dt_bf) {
    int idx = blockIdx.x * 256 + threadIdx.x;           // < ROWS_X*64
    int row = idx >> 6, col = idx & 63;
    dt_bf[idx] = f2bf(xdbl[(size_t)row * 96 + col]);
}

// ---------------- LayerNorm (+optional FiLM) -> bf16 ----------------
template<bool FILM>
__global__ __launch_bounds__(256) void ln_bf16_kernel(const float* __restrict__ X,
                                                      const float* __restrict__ w,
                                                      const float* __restrict__ bvec,
                                                      const float* __restrict__ gb,
                                                      u16* __restrict__ out) {
    __shared__ float red[256];
    const int row = blockIdx.x;
    const int tid = threadIdx.x;
    const float* xr = X + (size_t)row * D_MODEL;
    float v[4], s = 0.f, sq = 0.f;
#pragma unroll
    for (int i = 0; i < 4; ++i) { v[i] = xr[tid + i * 256]; s += v[i]; sq += v[i] * v[i]; }
    red[tid] = s; __syncthreads();
    for (int off = 128; off; off >>= 1) { if (tid < off) red[tid] += red[tid + off]; __syncthreads(); }
    float mean = red[0] * (1.f / D_MODEL); __syncthreads();
    red[tid] = sq; __syncthreads();
    for (int off = 128; off; off >>= 1) { if (tid < off) red[tid] += red[tid + off]; __syncthreads(); }
    float var = red[0] * (1.f / D_MODEL) - mean * mean;
    float rs = rsqrtf(var + 1e-5f);
    const int batch = row / SEQ_L;
#pragma unroll
    for (int i = 0; i < 4; ++i) {
        int c = tid + i * 256;
        float y = (v[i] - mean) * rs * w[c] + bvec[c];
        if (FILM) y = y * gb[(size_t)batch * 2048 + c] + gb[(size_t)batch * 2048 + 1024 + c];
        out[(size_t)row * D_MODEL + c] = f2bf(y);
    }
}

// ---------------- generic WMMA bf16 GEMM: D = epi(A @ W^T + bias) (+res) ----
// A: [M,K] bf16, W: [N,K] bf16, M%128==0, K%32==0, N guarded.
// Double-buffered async-to-LDS pipeline: stage k0+32 while computing k0.
// EPI: 0=none 2=gelu(exact) 3=softplus   OMODE: 0=f32 1=bf16 2=bf16 transposed
template<int EPI, int OMODE>
__global__ __launch_bounds__(256) void gemm_bf16_wmma(const u16* __restrict__ A,
                                                      const u16* __restrict__ W,
                                                      const float* __restrict__ bias,
                                                      const float* __restrict__ res,
                                                      void* __restrict__ outp,
                                                      int M, int N, int K) {
    __shared__ u16 sA[2][128 * 40];   // 32-col tile, stride 40 (80B, 16B aligned)
    __shared__ u16 sB[2][64 * 40];
    const int tid  = threadIdx.x;
    const int m0   = blockIdx.y * 128;
    const int n0   = blockIdx.x * 64;
    const int lane = tid & 31;
    const int wave = tid >> 5;
    const int wm = wave >> 1, wn = wave & 1;   // 4x2 wave grid, wave tile 32x32
    const int half = lane >> 4;
    const int lr   = lane & 15;

    f32x8 acc[2][2];
#pragma unroll
    for (int mi = 0; mi < 2; ++mi)
#pragma unroll
        for (int nj = 0; nj < 2; ++nj)
            acc[mi][nj] = (f32x8){0.f,0.f,0.f,0.f,0.f,0.f,0.f,0.f};

    const int arow = tid >> 2;            // 0..63
    const int acol = (tid & 3) * 8;       // 0,8,16,24
    // per-thread source rows (W row clamped; garbage cols land only in outputs
    // the epilogue discards via ng<N)
    const u16* gA0 = A + (size_t)(m0 + arow) * K + acol;
    const u16* gA1 = A + (size_t)(m0 + arow + 64) * K + acol;
    int wr = n0 + arow; if (wr >= N) wr = N - 1;
    const u16* gW  = W + (size_t)wr * K + acol;

    auto issue = [&](int buf, int k0) {
        ASYNC_B128(&sA[buf][arow * 40 + acol],        gA0 + k0);
        ASYNC_B128(&sA[buf][(arow + 64) * 40 + acol], gA1 + k0);
        ASYNC_B128(&sB[buf][arow * 40 + acol],        gW  + k0);
    };

    issue(0, 0);
    for (int k0 = 0; k0 < K; k0 += 32) {
        const int cur = (k0 >> 5) & 1;
        __syncthreads();                                // prior reads of buf cur^1 done
        if (k0 + 32 < K) { issue(cur ^ 1, k0 + 32); WAIT_ASYNCCNT(3); }
        else             { WAIT_ASYNCCNT(0); }
        __syncthreads();                                // stage `cur` visible to all waves

        FragBF a[2], b[2];
#pragma unroll
        for (int mi = 0; mi < 2; ++mi) {
            int r = wm * 32 + mi * 16 + lr;          // A layout: contiguous k = 16h..16h+15
            a[mi].q[0] = *(const int4*)&sA[cur][r * 40 + 16 * half];
            a[mi].q[1] = *(const int4*)&sA[cur][r * 40 + 16 * half + 8];
        }
#pragma unroll
        for (int nj = 0; nj < 2; ++nj) {
            int c = wn * 32 + nj * 16 + lr;          // B layout: contiguous k = 16h..16h+15
            b[nj].q[0] = *(const int4*)&sB[cur][c * 40 + 16 * half];
            b[nj].q[1] = *(const int4*)&sB[cur][c * 40 + 16 * half + 8];
        }
#pragma unroll
        for (int mi = 0; mi < 2; ++mi)
#pragma unroll
            for (int nj = 0; nj < 2; ++nj)
                acc[mi][nj] = __builtin_amdgcn_wmma_f32_16x16x32_bf16(
                    false, a[mi].v, false, b[nj].v, (short)0, acc[mi][nj], false, false);
    }

    // epilogue: C layout -> m = r + 8*half, n = lane&15
#pragma unroll
    for (int mi = 0; mi < 2; ++mi)
#pragma unroll
        for (int nj = 0; nj < 2; ++nj) {
            int ng = n0 + wn * 32 + nj * 16 + lr;
            if (ng < N) {
                float bv = bias ? bias[ng] : 0.f;
#pragma unroll
                for (int r = 0; r < 8; ++r) {
                    int mg = m0 + wm * 32 + mi * 16 + r + 8 * half;
                    float v = acc[mi][nj][r] + bv;
                    if (EPI == 2) v = 0.5f * v * (1.f + erff(v * 0.70710678118654752f));
                    if (EPI == 3) v = (v > 20.f) ? v : log1pf(__expf(v));
                    if (res) v += res[(size_t)mg * N + ng];
                    if (OMODE == 0)      ((float*)outp)[(size_t)mg * N + ng] = v;
                    else if (OMODE == 1) ((u16*)outp)[(size_t)mg * N + ng]   = f2bf(v);
                    else                 ((u16*)outp)[(size_t)ng * M + mg]   = f2bf(v); // transposed
                }
            }
        }
}

// ---------------- depthwise causal conv + SiLU, plus silu(z) ----------------
__global__ __launch_bounds__(256) void conv_silu_kernel(const float* __restrict__ xz,
                                                        const float* __restrict__ conv_w,
                                                        const float* __restrict__ conv_b,
                                                        float* __restrict__ u_f32,
                                                        u16*   __restrict__ u_bf,
                                                        float* __restrict__ sz) {
    int idx = blockIdx.x * 256 + threadIdx.x;     // < ROWS_X * D_INNER
    int c  = idx & (D_INNER - 1);
    int bl = idx >> 11;
    int l  = bl & (SEQ_L - 1);
    int b  = bl >> 9;
    float accv = conv_b[c];
#pragma unroll
    for (int j = 0; j < D_CONV; ++j) {
        int lt = l - (D_CONV - 1) + j;
        if (lt >= 0)
            accv += conv_w[c * D_CONV + j] * xz[(size_t)(b * SEQ_L + lt) * 4096 + c];
    }
    float us = accv / (1.f + __expf(-accv));
    u_f32[idx] = us;
    u_bf[idx]  = f2bf(us);
    float z = xz[(size_t)bl * 4096 + 2048 + c];
    sz[idx] = z / (1.f + __expf(-z));
}

// ---------------- selective scan (sequential over L, parallel over channels) --
__global__ __launch_bounds__(512) void scan_kernel(const float* __restrict__ xdbl,
                                                   const float* __restrict__ delta,
                                                   const float* __restrict__ u_f32,
                                                   const float* __restrict__ sz,
                                                   const float* __restrict__ a_log,
                                                   const float* __restrict__ d_skip,
                                                   u16*   __restrict__ y_bf,
                                                   float* __restrict__ state_out) {
    __shared__ float sBC[32];
    const int tid = threadIdx.x;
    const int chunk = blockIdx.x;     // 0..3
    const int b = blockIdx.y;         // 0..1
    const int d = chunk * 512 + tid;  // channel 0..2047
    float a[D_STATE], hc[D_STATE];
#pragma unroll
    for (int s = 0; s < D_STATE; ++s) { a[s] = -__expf(a_log[(size_t)d * D_STATE + s]); hc[s] = 0.f; }
    const float dsk = d_skip[d];
    for (int t = 0; t < SEQ_L; ++t) {
        __syncthreads();
        if (tid < 32) sBC[tid] = xdbl[(size_t)(b * SEQ_L + t) * 96 + 64 + tid];
        __syncthreads();
        size_t idx = (size_t)(b * SEQ_L + t) * D_INNER + d;
        float dl = delta[idx], uu = u_f32[idx];
        float du = dl * uu;
        float accv = 0.f;
#pragma unroll
        for (int s = 0; s < D_STATE; ++s) {
            hc[s] = __expf(dl * a[s]) * hc[s] + du * sBC[s];
            accv += hc[s] * sBC[16 + s];
        }
        float y = (accv + uu * dsk) * sz[idx];
        y_bf[idx] = f2bf(y);
    }
#pragma unroll
    for (int s = 0; s < D_STATE; ++s)
        state_out[((size_t)b * D_INNER + d) * D_STATE + s] = hc[s];
}

// ---------------- cross attention: one wave per (b, head, 16-query tile) ------
__global__ __launch_bounds__(32) void attn_kernel(const u16* __restrict__ q_bf,
                                                  const u16* __restrict__ k_bf,
                                                  const u16* __restrict__ vT_bf,
                                                  const unsigned char* __restrict__ mask,
                                                  u16* __restrict__ ao_bf) {
    __shared__ float S[16][SEQ_T];     // 16KB
    __shared__ u16   P[16][SEQ_T];     // 8KB
    const int lane = threadIdx.x;
    const int half = lane >> 4, lr = lane & 15;
    const int l0 = blockIdx.x * 16;
    const int h  = blockIdx.y;
    const int b  = blockIdx.z;

    FragBF aq[2];
    const u16* qrow = q_bf + (size_t)(b * SEQ_L + l0 + lr) * D_MODEL + h * HEAD_D;
#pragma unroll
    for (int kd = 0; kd < 2; ++kd) {
        aq[kd].q[0] = *(const int4*)(qrow + kd * 32 + 16 * half);
        aq[kd].q[1] = *(const int4*)(qrow + kd * 32 + 16 * half + 8);
    }
    // scores S = Q K^T * 1/8, masked
    for (int n0 = 0; n0 < SEQ_T; n0 += 16) {
        FragBF bk[2];
        const u16* krow = k_bf + (size_t)(b * SEQ_T + n0 + lr) * D_MODEL + h * HEAD_D;
#pragma unroll
        for (int kd = 0; kd < 2; ++kd) {
            bk[kd].q[0] = *(const int4*)(krow + kd * 32 + 16 * half);
            bk[kd].q[1] = *(const int4*)(krow + kd * 32 + 16 * half + 8);
        }
        f32x8 accs = (f32x8){0.f,0.f,0.f,0.f,0.f,0.f,0.f,0.f};
        accs = __builtin_amdgcn_wmma_f32_16x16x32_bf16(false, aq[0].v, false, bk[0].v, (short)0, accs, false, false);
        accs = __builtin_amdgcn_wmma_f32_16x16x32_bf16(false, aq[1].v, false, bk[1].v, (short)0, accs, false, false);
        int nc = n0 + lr;
        float mval = mask[(size_t)b * SEQ_T + nc] ? 0.f : -1e30f;
#pragma unroll
        for (int r = 0; r < 8; ++r)
            S[r + 8 * half][nc] = accs[r] * 0.125f + mval;
    }
    __syncthreads();
    if (lane < 16) {
        float mx = -3.4e38f;
        for (int j = 0; j < SEQ_T; ++j) mx = fmaxf(mx, S[lane][j]);
        float s = 0.f;
        for (int j = 0; j < SEQ_T; ++j) { float e = __expf(S[lane][j] - mx); S[lane][j] = e; s += e; }
        float inv = 1.f / s;
        for (int j = 0; j < SEQ_T; ++j) P[lane][j] = f2bf(S[lane][j] * inv);
    }
    __syncthreads();
    // out = P @ V   (V read transposed: vT[d][b*256 + kpos])
    f32x8 o[4];
#pragma unroll
    for (int j = 0; j < 4; ++j) o[j] = (f32x8){0.f,0.f,0.f,0.f,0.f,0.f,0.f,0.f};
    for (int kd0 = 0; kd0 < SEQ_T; kd0 += 32) {
        FragBF ap;
        ap.q[0] = *(const int4*)&P[lr][kd0 + 16 * half];
        ap.q[1] = *(const int4*)&P[lr][kd0 + 16 * half + 8];
#pragma unroll
        for (int j = 0; j < 4; ++j) {
            FragBF bv;
            const u16* vrow = vT_bf + (size_t)(h * HEAD_D + j * 16 + lr) * ROWS_T
                              + b * SEQ_T + kd0 + 16 * half;
            bv.q[0] = *(const int4*)(vrow);
            bv.q[1] = *(const int4*)(vrow + 8);
            o[j] = __builtin_amdgcn_wmma_f32_16x16x32_bf16(false, ap.v, false, bv.v, (short)0, o[j], false, false);
        }
    }
#pragma unroll
    for (int j = 0; j < 4; ++j)
#pragma unroll
        for (int r = 0; r < 8; ++r)
            ao_bf[(size_t)(b * SEQ_L + l0 + r + 8 * half) * D_MODEL + h * HEAD_D + j * 16 + lr] = f2bf(o[j][r]);
}

// ---------------- style FiLM: gb = tanh(z_style @ w_style^T + b_style) --------
__global__ __launch_bounds__(256) void style_kernel(const float* __restrict__ z_style,
                                                    const float* __restrict__ w_style,
                                                    const float* __restrict__ b_style,
                                                    float* __restrict__ gb) {
    int o = blockIdx.x * 256 + threadIdx.x;   // < BATCH*2048
    int b = o >> 11, j = o & 2047;
    float accv = b_style[j];
    const float* wr = w_style + (size_t)j * D_STYLE;
    const float* zr = z_style + (size_t)b * D_STYLE;
    for (int r = 0; r < D_STYLE; ++r) accv += zr[r] * wr[r];
    gb[o] = tanhf(accv);
}

// =======================================================================
extern "C" void kernel_launch(void* const* d_in, const int* in_sizes, int n_in,
                              void* d_out, int out_size, void* d_ws, size_t ws_size,
                              hipStream_t stream) {
    (void)in_sizes; (void)n_in; (void)out_size; (void)ws_size;
    const float* x_in   = (const float*)d_in[0];
    const float* th     = (const float*)d_in[1];
    const float* zsty   = (const float*)d_in[2];
    const unsigned char* tmask = (const unsigned char*)d_in[3];
    const float* ln1w = (const float*)d_in[4],  *ln1b = (const float*)d_in[5];
    const float* ln2w = (const float*)d_in[6],  *ln2b = (const float*)d_in[7];
    const float* ln3w = (const float*)d_in[8],  *ln3b = (const float*)d_in[9];
    const float* w_in = (const float*)d_in[10], *b_in = (const float*)d_in[11];
    const float* convw = (const float*)d_in[12], *convb = (const float*)d_in[13];
    const float* w_xproj = (const float*)d_in[14];
    const float* w_dt = (const float*)d_in[15], *b_dt = (const float*)d_in[16];
    const float* a_log = (const float*)d_in[17], *d_skip = (const float*)d_in[18];
    const float* w_mout = (const float*)d_in[19], *b_mout = (const float*)d_in[20];
    const float* wq = (const float*)d_in[21], *bq = (const float*)d_in[22];
    const float* wk = (const float*)d_in[23], *bk = (const float*)d_in[24];
    const float* wv = (const float*)d_in[25], *bv = (const float*)d_in[26];
    const float* wao = (const float*)d_in[27], *bao = (const float*)d_in[28];
    const float* wff1 = (const float*)d_in[29], *bff1 = (const float*)d_in[30];
    const float* wff2 = (const float*)d_in[31], *bff2 = (const float*)d_in[32];
    const float* wsty = (const float*)d_in[33], *bsty = (const float*)d_in[34];

    float* out_x     = (float*)d_out;                      // [B,L,D]  residual stream
    float* out_state = (float*)d_out + (size_t)ROWS_X * D_MODEL;  // [B,D_INNER,D_STATE]

    // ---- workspace carve-out ----
    size_t off = 0;
    auto aU16 = [&](size_t n) { u16* p = (u16*)((char*)d_ws + off); off += ((n * 2 + 255) / 256) * 256; return p; };
    auto aF32 = [&](size_t n) { float* p = (float*)((char*)d_ws + off); off += ((n * 4 + 255) / 256) * 256; return p; };

    u16* win_bf = aU16((size_t)4096 * 1024);
    u16* wxp_bf = aU16((size_t)96 * 2048);
    u16* wdt_bf = aU16((size_t)2048 * 64);
    u16* wmo_bf = aU16((size_t)1024 * 2048);
    u16* wq_bf  = aU16((size_t)1024 * 1024);
    u16* wk_bf  = aU16((size_t)1024 * 1024);
    u16* wv_bf  = aU16((size_t)1024 * 1024);
    u16* wao_bf = aU16((size_t)1024 * 1024);
    u16* wf1_bf = aU16((size_t)4096 * 1024);
    u16* wf2_bf = aU16((size_t)1024 * 4096);
    u16* h1_bf  = aU16((size_t)ROWS_X * D_MODEL);
    u16* u_bf   = aU16((size_t)ROWS_X * D_INNER);
    u16* dt_bf  = aU16((size_t)ROWS_X * DT_RANK);
    u16* y_bf   = aU16((size_t)ROWS_X * D_INNER);
    u16* h2_bf  = aU16((size_t)ROWS_X * D_MODEL);
    u16* th_bf  = aU16((size_t)ROWS_T * D_MODEL);
    u16* q_bf   = aU16((size_t)ROWS_X * D_MODEL);
    u16* k_bf   = aU16((size_t)ROWS_T * D_MODEL);
    u16* vT_bf  = aU16((size_t)D_MODEL * ROWS_T);
    u16* ao_bf  = aU16((size_t)ROWS_X * D_MODEL);
    u16* h3_bf  = aU16((size_t)ROWS_X * D_MODEL);
    u16* ff1_bf = aU16((size_t)ROWS_X * D_FF);
    float* xz    = aF32((size_t)ROWS_X * 4096);
    float* u_f32 = aF32((size_t)ROWS_X * D_INNER);
    float* sz    = aF32((size_t)ROWS_X * D_INNER);
    float* xdbl  = aF32((size_t)ROWS_X * 96);
    float* delta = aF32((size_t)ROWS_X * D_INNER);
    float* gb    = aF32((size_t)BATCH * 2048);

    // ---- weight / input bf16 conversion ----
    auto cvt = [&](const float* s, u16* d, size_t n) {
        cvt_kernel<<<1024, 256, 0, stream>>>(s, d, (int)n);
    };
    cvt(w_in, win_bf, (size_t)4096 * 1024);
    cvt(w_xproj, wxp_bf, (size_t)96 * 2048);
    cvt(w_dt, wdt_bf, (size_t)2048 * 64);
    cvt(w_mout, wmo_bf, (size_t)1024 * 2048);
    cvt(wq, wq_bf, (size_t)1024 * 1024);
    cvt(wk, wk_bf, (size_t)1024 * 1024);
    cvt(wv, wv_bf, (size_t)1024 * 1024);
    cvt(wao, wao_bf, (size_t)1024 * 1024);
    cvt(wff1, wf1_bf, (size_t)4096 * 1024);
    cvt(wff2, wf2_bf, (size_t)1024 * 4096);
    cvt(th, th_bf, (size_t)ROWS_T * D_MODEL);

    // ---- Mamba branch ----
    ln_bf16_kernel<false><<<ROWS_X, 256, 0, stream>>>(x_in, ln1w, ln1b, nullptr, h1_bf);
    gemm_bf16_wmma<0, 0><<<dim3(4096 / 64, ROWS_X / 128), 256, 0, stream>>>(
        h1_bf, win_bf, b_in, nullptr, xz, ROWS_X, 4096, 1024);
    conv_silu_kernel<<<(ROWS_X * D_INNER) / 256, 256, 0, stream>>>(xz, convw, convb, u_f32, u_bf, sz);
    gemm_bf16_wmma<0, 0><<<dim3(2, ROWS_X / 128), 256, 0, stream>>>(
        u_bf, wxp_bf, nullptr, nullptr, xdbl, ROWS_X, 96, 2048);
    dtcvt_kernel<<<(ROWS_X * DT_RANK) / 256, 256, 0, stream>>>(xdbl, dt_bf);
    gemm_bf16_wmma<3, 0><<<dim3(2048 / 64, ROWS_X / 128), 256, 0, stream>>>(
        dt_bf, wdt_bf, b_dt, nullptr, delta, ROWS_X, 2048, 64);
    scan_kernel<<<dim3(4, 2), 512, 0, stream>>>(xdbl, delta, u_f32, sz, a_log, d_skip, y_bf, out_state);
    gemm_bf16_wmma<0, 0><<<dim3(1024 / 64, ROWS_X / 128), 256, 0, stream>>>(
        y_bf, wmo_bf, b_mout, x_in, out_x, ROWS_X, 1024, 2048);   // x1 = x + mamba

    // ---- cross attention ----
    ln_bf16_kernel<false><<<ROWS_X, 256, 0, stream>>>(out_x, ln2w, ln2b, nullptr, h2_bf);
    gemm_bf16_wmma<0, 1><<<dim3(1024 / 64, ROWS_X / 128), 256, 0, stream>>>(
        h2_bf, wq_bf, bq, nullptr, q_bf, ROWS_X, 1024, 1024);
    gemm_bf16_wmma<0, 1><<<dim3(1024 / 64, ROWS_T / 128), 256, 0, stream>>>(
        th_bf, wk_bf, bk, nullptr, k_bf, ROWS_T, 1024, 1024);
    gemm_bf16_wmma<0, 2><<<dim3(1024 / 64, ROWS_T / 128), 256, 0, stream>>>(
        th_bf, wv_bf, bv, nullptr, vT_bf, ROWS_T, 1024, 1024);    // V^T for PV WMMA
    attn_kernel<<<dim3(SEQ_L / 16, N_HEADS, BATCH), 32, 0, stream>>>(q_bf, k_bf, vT_bf, tmask, ao_bf);
    gemm_bf16_wmma<0, 0><<<dim3(1024 / 64, ROWS_X / 128), 256, 0, stream>>>(
        ao_bf, wao_bf, bao, out_x, out_x, ROWS_X, 1024, 1024);    // x2 = x1 + attn

    // ---- FiLM-modulated FFN ----
    style_kernel<<<(BATCH * 2048) / 256, 256, 0, stream>>>(zsty, wsty, bsty, gb);
    ln_bf16_kernel<true><<<ROWS_X, 256, 0, stream>>>(out_x, ln3w, ln3b, gb, h3_bf);
    gemm_bf16_wmma<2, 1><<<dim3(D_FF / 64, ROWS_X / 128), 256, 0, stream>>>(
        h3_bf, wf1_bf, bff1, nullptr, ff1_bf, ROWS_X, D_FF, 1024);
    gemm_bf16_wmma<0, 0><<<dim3(1024 / 64, ROWS_X / 128), 256, 0, stream>>>(
        ff1_bf, wf2_bf, bff2, out_x, out_x, ROWS_X, 1024, D_FF);  // x3 = x2 + ffn
}
